// CausalSelfAttention_17729624998076
// MI455X (gfx1250) — matrix operand order repaired and checked by reading, not compiled, not run on hardware
//
#include <hip/hip_runtime.h>
#include <hip/hip_bf16.h>

typedef __attribute__((ext_vector_type(16))) _Float16 v16h;
typedef __attribute__((ext_vector_type(8)))  _Float16 v8h;
typedef __attribute__((ext_vector_type(8)))  float    v8f;

#define Bb 4
#define Tt 2048
#define Cc 1024
#define Hh 16
#define Dd 64
#define Mtot (Bb*Tt)     // 8192
#define ASTR 40          // padded LDS stride (halves) for GEMM tiles (80B = 20 banks)
#define KVSTR 72         // padded LDS stride (halves) for K/V/P tiles (144B)

static __device__ inline v16h join16(v8h lo, v8h hi) {
    return __builtin_shufflevector(lo, hi, 0,1,2,3,4,5,6,7,8,9,10,11,12,13,14,15);
}

// ---------------------------------------------------------------------------
// Kernel 1: QKV projection. x(8192x1024) @ Wqkv(1024x3072) + b -> scatter to
// q[b,h,t,d], k[b,h,t,d] (f16) and vT[b,h,d,t] (f16, transposed for P@V).
// 256 threads (8 waves), 128x128 block tile, wave tile 32x64 (2x4 WMMA frags).
// ---------------------------------------------------------------------------
__global__ __launch_bounds__(256) void qkv_gemm_kernel(
    const float* __restrict__ x, const float* __restrict__ Wqkv,
    const float* __restrict__ bqkv,
    _Float16* __restrict__ qh, _Float16* __restrict__ kh,
    _Float16* __restrict__ vT)
{
    __shared__ _Float16 ldsA[128 * ASTR];   // [m][k]
    __shared__ _Float16 ldsB[128 * ASTR];   // [n][k] (transposed W tile)
    const int tid  = threadIdx.x;
    const int lane = tid & 31;
    const int wid  = tid >> 5;
    const int tile_n = blockIdx.x * 128;
    const int tile_m = blockIdx.y * 128;
    const int K = Cc, N = 3 * Cc;

    v8f acc[2][4] = {};
    const int wrow = (wid >> 1) * 32;       // 4 wave-rows of 32
    const int wcol = (wid & 1) * 64;        // 2 wave-cols of 64

    for (int k0 = 0; k0 < K; k0 += 32) {
        // stage A tile 128x32 (f32 -> f16)
        #pragma unroll
        for (int it = 0; it < 4; ++it) {
            int c   = tid + it * 256;
            int row = c >> 3;
            int col = (c & 7) << 2;
            float4 f = *(const float4*)(x + (size_t)(tile_m + row) * K + k0 + col);
            _Float16* dst = &ldsA[row * ASTR + col];
            dst[0] = (_Float16)f.x; dst[1] = (_Float16)f.y;
            dst[2] = (_Float16)f.z; dst[3] = (_Float16)f.w;
        }
        // stage W tile 32x128 transposed -> ldsB[n][k]
        #pragma unroll
        for (int it = 0; it < 4; ++it) {
            int c    = tid + it * 256;
            int krow = c >> 5;
            int col  = (c & 31) << 2;
            float4 f = *(const float4*)(Wqkv + (size_t)(k0 + krow) * N + tile_n + col);
            ldsB[(col + 0) * ASTR + krow] = (_Float16)f.x;
            ldsB[(col + 1) * ASTR + krow] = (_Float16)f.y;
            ldsB[(col + 2) * ASTR + krow] = (_Float16)f.z;
            ldsB[(col + 3) * ASTR + krow] = (_Float16)f.w;
        }
        __syncthreads();

        v16h afrag[2], bfrag[4];
        const int ah  = (lane < 16) ? 0 : 8;    // A: K half-offset per lane half
        const int bkb = (lane < 16) ? 0 : 8;    // B: K base per lane half
        #pragma unroll
        for (int i = 0; i < 2; ++i) {
            const _Float16* ap = &ldsA[(wrow + i * 16 + (lane & 15)) * ASTR + ah];
            afrag[i] = join16(*(const v8h*)ap, *(const v8h*)(ap + 16));
        }
        #pragma unroll
        for (int j = 0; j < 4; ++j) {
            const _Float16* bp = &ldsB[(wcol + j * 16 + (lane & 15)) * ASTR + bkb];
            bfrag[j] = join16(*(const v8h*)bp, *(const v8h*)(bp + 16));
        }
        #pragma unroll
        for (int i = 0; i < 2; ++i)
            #pragma unroll
            for (int j = 0; j < 4; ++j)
                acc[i][j] = __builtin_amdgcn_wmma_f32_16x16x32_f16(
                    false, afrag[i], false, bfrag[j], (short)0, acc[i][j], false, false);
        __syncthreads();
    }

    // epilogue: bias + scatter; 'which' is block-uniform (tile_n 128-aligned)
    const int which = tile_n >> 10;
    const int hil = (lane >= 16) ? 8 : 0;
    if (which == 2) {
        #pragma unroll
        for (int i = 0; i < 2; ++i)
            #pragma unroll
            for (int j = 0; j < 4; ++j)
                #pragma unroll
                for (int r = 0; r < 8; ++r) {
                    int m = tile_m + wrow + i * 16 + r + hil;
                    int n = tile_n + wcol + j * 16 + (lane & 15);
                    float v = acc[i][j][r] + bqkv[n];
                    int b = m >> 11, t = m & (Tt - 1);
                    int cc = n & (Cc - 1), h = cc >> 6, d = cc & (Dd - 1);
                    vT[(((size_t)b * Hh + h) * Dd + d) * Tt + t] = (_Float16)v;
                }
    } else {
        _Float16* dst = which ? kh : qh;
        #pragma unroll
        for (int i = 0; i < 2; ++i)
            #pragma unroll
            for (int j = 0; j < 4; ++j)
                #pragma unroll
                for (int r = 0; r < 8; ++r) {
                    int m = tile_m + wrow + i * 16 + r + hil;
                    int n = tile_n + wcol + j * 16 + (lane & 15);
                    float v = acc[i][j][r] + bqkv[n];
                    int b = m >> 11, t = m & (Tt - 1);
                    int cc = n & (Cc - 1), h = cc >> 6, d = cc & (Dd - 1);
                    dst[(((size_t)b * Hh + h) * Tt + t) * Dd + d] = (_Float16)v;
                }
    }
}

// ---------------------------------------------------------------------------
// Async copy of one 64-row x 128-byte tile (64x64 f16) global -> LDS,
// padded LDS row stride. 128 threads, 4 x b128 chunks per thread.
// ---------------------------------------------------------------------------
static __device__ inline void async_tile_64x64(const _Float16* g, int gstride,
                                               _Float16* l, int tid)
{
    #pragma unroll
    for (int it = 0; it < 4; ++it) {
        int c   = tid + it * 128;
        int row = c >> 3;
        int ch  = (c & 7) * 8;                        // halves
        const _Float16* ga = g + (size_t)row * gstride + ch;
        unsigned la = (unsigned)(uintptr_t)(l + row * KVSTR + ch);
        asm volatile("global_load_async_to_lds_b128 %0, %1, off"
                     :: "v"(la), "v"((unsigned long long)(uintptr_t)ga)
                     : "memory");
    }
}

// ---------------------------------------------------------------------------
// Kernel 2: flash attention. grid = (T/64, B*H); block = 128 (4 waves).
// K/V tiles double-buffered in LDS via async-to-LDS; shared by all 4 waves.
// ---------------------------------------------------------------------------
__global__ __launch_bounds__(128) void attn_kernel(
    const _Float16* __restrict__ qh, const _Float16* __restrict__ kh,
    const _Float16* __restrict__ vT, const int* __restrict__ attn_mask,
    _Float16* __restrict__ yh)
{
    __shared__ _Float16 ldsK[2][64 * KVSTR];
    __shared__ _Float16 ldsV[2][64 * KVSTR];
    __shared__ _Float16 ldsP[4][16 * KVSTR];
    const int tid  = threadIdx.x;
    const int lane = tid & 31;
    const int wid  = tid >> 5;
    const int bh   = blockIdx.y;
    const int b    = bh >> 4;
    const int h    = bh & (Hh - 1);
    const int qtile = blockIdx.x;
    const int row0  = qtile * 64 + wid * 16;
    const int hilane = (lane >> 4) & 1;
    const float cexp = 1.4426950408889634f * 0.125f;   // log2(e)/sqrt(D)

    const _Float16* kbase = kh + (size_t)bh * Tt * Dd;         // [t][d]
    const _Float16* vbase = vT + (size_t)bh * Dd * Tt;         // [d][t]

    // Q fragments (K = D = 64 -> two k=32 fragments), ISA A-layout
    v16h aq[2];
    {
        const _Float16* qrow = qh + ((size_t)bh * Tt + row0 + (lane & 15)) * Dd;
        const int ah = (lane < 16) ? 0 : 8;
        #pragma unroll
        for (int kk = 0; kk < 2; ++kk)
            aq[kk] = join16(*(const v8h*)(qrow + kk * 32 + ah),
                            *(const v8h*)(qrow + kk * 32 + ah + 16));
    }

    v8f of[4] = {};
    float mrun[8], lrun[8];
    #pragma unroll
    for (int r = 0; r < 8; ++r) { mrun[r] = -1e30f; lrun[r] = 0.0f; }

    // prologue: stream first K/V tile into buffer 0
    async_tile_64x64(kbase, Dd, &ldsK[0][0], tid);
    async_tile_64x64(vbase, Tt, &ldsV[0][0], tid);

    for (int st = 0; st <= qtile; ++st) {
        const int sb  = st * 64;
        const int cur = st & 1;
        asm volatile("s_wait_asynccnt 0x0" ::: "memory");
        __syncthreads();
        if (st < qtile) {   // prefetch next tile into the other buffer
            const int sb2 = sb + 64;
            async_tile_64x64(kbase + (size_t)sb2 * Dd, Dd, &ldsK[1 - cur][0], tid);
            async_tile_64x64(vbase + sb2,              Tt, &ldsV[1 - cur][0], tid);
        }
        const _Float16* curK = &ldsK[cur][0];
        const _Float16* curV = &ldsV[cur][0];

        // S = Q K^T (B frag = contiguous d-run of LDS K tile)
        v8f sf[4];
        int pad[4];
        #pragma unroll
        for (int jn = 0; jn < 4; ++jn) {
            const int sl = jn * 16 + (lane & 15);
            pad[jn] = attn_mask[(size_t)b * Tt + sb + sl];
            v8f s = {};
            #pragma unroll
            for (int kk = 0; kk < 2; ++kk) {
                const _Float16* kp = &curK[sl * KVSTR + kk * 32 + ((lane < 16) ? 0 : 8)];
                v16h bfr = join16(*(const v8h*)kp, *(const v8h*)(kp + 16));
                s = __builtin_amdgcn_wmma_f32_16x16x32_f16(
                        false, aq[kk], false, bfr, (short)0, s, false, false);
            }
            sf[jn] = s;
        }
        // causal + pad masking in C-layout
        if (st == qtile) {
            #pragma unroll
            for (int jn = 0; jn < 4; ++jn) {
                int scol = sb + jn * 16 + (lane & 15);
                #pragma unroll
                for (int r = 0; r < 8; ++r) {
                    int trow = row0 + r + hilane * 8;
                    if (scol > trow || !pad[jn]) sf[jn][r] = -1e30f;
                }
            }
        } else {
            #pragma unroll
            for (int jn = 0; jn < 4; ++jn)
                if (!pad[jn])
                    #pragma unroll
                    for (int r = 0; r < 8; ++r) sf[jn][r] = -1e30f;
        }
        // online softmax (row reductions across the 16-lane half, wave32)
        float alpha[8];
        #pragma unroll
        for (int r = 0; r < 8; ++r) {
            float mx = fmaxf(fmaxf(sf[0][r], sf[1][r]), fmaxf(sf[2][r], sf[3][r]));
            #pragma unroll
            for (int off = 8; off >= 1; off >>= 1)
                mx = fmaxf(mx, __shfl_xor(mx, off, 32));
            float mnew = fmaxf(mrun[r], mx);
            alpha[r] = exp2f((mrun[r] - mnew) * cexp);
            mrun[r] = mnew;
            float ls = 0.0f;
            #pragma unroll
            for (int jn = 0; jn < 4; ++jn) {
                float p = exp2f((sf[jn][r] - mnew) * cexp);
                sf[jn][r] = p;
                ls += p;
            }
            #pragma unroll
            for (int off = 8; off >= 1; off >>= 1)
                ls += __shfl_xor(ls, off, 32);
            lrun[r] = lrun[r] * alpha[r] + ls;
        }
        #pragma unroll
        for (int dj = 0; dj < 4; ++dj)
            #pragma unroll
            for (int r = 0; r < 8; ++r) of[dj][r] *= alpha[r];

        // relay P through per-wave LDS (C-layout -> A-layout); intra-wave only
        _Float16* pbuf = &ldsP[wid][0];
        #pragma unroll
        for (int jn = 0; jn < 4; ++jn) {
            int col = jn * 16 + (lane & 15);
            #pragma unroll
            for (int r = 0; r < 8; ++r)
                pbuf[(r + hilane * 8) * KVSTR + col] = (_Float16)sf[jn][r];
        }
        asm volatile("s_wait_dscnt 0x0" ::: "memory");
        v16h ap[2];
        {
            const _Float16* prow = pbuf + (lane & 15) * KVSTR;
            const int ah = (lane < 16) ? 0 : 8;
            #pragma unroll
            for (int kk = 0; kk < 2; ++kk)
                ap[kk] = join16(*(const v8h*)(prow + kk * 32 + ah),
                                *(const v8h*)(prow + kk * 32 + ah + 16));
        }
        // O += P V (B frag = contiguous s-run of LDS V tile)
        #pragma unroll
        for (int dj = 0; dj < 4; ++dj) {
            const _Float16* vrow = &curV[(dj * 16 + (lane & 15)) * KVSTR];
            #pragma unroll
            for (int kk = 0; kk < 2; ++kk) {
                const _Float16* vp = vrow + kk * 32 + ((lane < 16) ? 0 : 8);
                v16h bfr = join16(*(const v8h*)vp, *(const v8h*)(vp + 16));
                of[dj] = __builtin_amdgcn_wmma_f32_16x16x32_f16(
                             false, ap[kk], false, bfr, (short)0, of[dj], false, false);
            }
        }
    }

    // normalize, store Y as (b,t,h*64+d) f16
    #pragma unroll
    for (int dj = 0; dj < 4; ++dj) {
        int d = dj * 16 + (lane & 15);
        #pragma unroll
        for (int r = 0; r < 8; ++r) {
            int trow = row0 + r + hilane * 8;
            float l = lrun[r];
            float o = (l > 0.0f) ? of[dj][r] / l : 0.0f;
            yh[((size_t)b * Tt + trow) * Cc + h * Dd + d] = (_Float16)o;
        }
    }
}

// ---------------------------------------------------------------------------
// Kernel 3: output projection. Y_f16(8192x1024) @ Wproj(1024x1024) + b -> f32.
// 256 threads, 128x128 tile; A fragments straight from global (Y is f16).
// ---------------------------------------------------------------------------
__global__ __launch_bounds__(256) void proj_gemm_kernel(
    const _Float16* __restrict__ yh, const float* __restrict__ Wp,
    const float* __restrict__ bp, float* __restrict__ out)
{
    __shared__ _Float16 ldsB[128 * ASTR];
    const int tid  = threadIdx.x;
    const int lane = tid & 31;
    const int wid  = tid >> 5;
    const int tile_n = blockIdx.x * 128;
    const int tile_m = blockIdx.y * 128;
    const int K = Cc, N = Cc;

    v8f acc[2][4] = {};
    const int wrow = (wid >> 1) * 32;
    const int wcol = (wid & 1) * 64;

    for (int k0 = 0; k0 < K; k0 += 32) {
        #pragma unroll
        for (int it = 0; it < 4; ++it) {
            int c    = tid + it * 256;
            int krow = c >> 5;
            int col  = (c & 31) << 2;
            float4 f = *(const float4*)(Wp + (size_t)(k0 + krow) * N + tile_n + col);
            ldsB[(col + 0) * ASTR + krow] = (_Float16)f.x;
            ldsB[(col + 1) * ASTR + krow] = (_Float16)f.y;
            ldsB[(col + 2) * ASTR + krow] = (_Float16)f.z;
            ldsB[(col + 3) * ASTR + krow] = (_Float16)f.w;
        }
        __syncthreads();

        v16h afrag[2], bfrag[4];
        const int ah  = (lane < 16) ? 0 : 8;
        const int bkb = (lane < 16) ? 0 : 8;
        #pragma unroll
        for (int i = 0; i < 2; ++i) {
            const _Float16* ap = yh + (size_t)(tile_m + wrow + i * 16 + (lane & 15)) * K + k0 + ah;
            afrag[i] = join16(*(const v8h*)ap, *(const v8h*)(ap + 16));
        }
        #pragma unroll
        for (int j = 0; j < 4; ++j) {
            const _Float16* bpp = &ldsB[(wcol + j * 16 + (lane & 15)) * ASTR + bkb];
            bfrag[j] = join16(*(const v8h*)bpp, *(const v8h*)(bpp + 16));
        }
        #pragma unroll
        for (int i = 0; i < 2; ++i)
            #pragma unroll
            for (int j = 0; j < 4; ++j)
                acc[i][j] = __builtin_amdgcn_wmma_f32_16x16x32_f16(
                    false, afrag[i], false, bfrag[j], (short)0, acc[i][j], false, false);
        __syncthreads();
    }

    const int hil = (lane >= 16) ? 8 : 0;
    #pragma unroll
    for (int i = 0; i < 2; ++i)
        #pragma unroll
        for (int j = 0; j < 4; ++j)
            #pragma unroll
            for (int r = 0; r < 8; ++r) {
                int m = tile_m + wrow + i * 16 + r + hil;
                int n = tile_n + wcol + j * 16 + (lane & 15);
                out[(size_t)m * N + n] = acc[i][j][r] + bp[n];
            }
}

// ---------------------------------------------------------------------------
extern "C" void kernel_launch(void* const* d_in, const int* in_sizes, int n_in,
                              void* d_out, int out_size, void* d_ws, size_t ws_size,
                              hipStream_t stream) {
    const float* x    = (const float*)d_in[0];
    const int*   am   = (const int*)  d_in[1];
    const float* Wqkv = (const float*)d_in[2];
    const float* bqkv = (const float*)d_in[3];
    const float* Wp   = (const float*)d_in[4];
    const float* bp   = (const float*)d_in[5];
    float* out = (float*)d_out;

    const size_t NQ = (size_t)Bb * Tt * Cc;   // 8,388,608 elements per tensor
    _Float16* qh = (_Float16*)d_ws;
    _Float16* kh = qh + NQ;
    _Float16* vT = kh + NQ;
    _Float16* yh = vT + NQ;                   // 64 MB total f16 workspace

    qkv_gemm_kernel<<<dim3(3 * Cc / 128, Mtot / 128), dim3(256), 0, stream>>>(
        x, Wqkv, bqkv, qh, kh, vT);
    attn_kernel<<<dim3(Tt / 64, Bb * Hh), dim3(128), 0, stream>>>(
        qh, kh, vT, am, yh);
    proj_gemm_kernel<<<dim3(Cc / 128, Mtot / 128), dim3(256), 0, stream>>>(
        yh, Wp, bp, out);
}
